// BOTNet_43293270343919
// MI455X (gfx1250) — matrix operand order repaired and checked by reading, compiled, not run on hardware
//
#include <hip/hip_runtime.h>

// ---------------------------------------------------------------------------
// BOTNet-style equivariant GNN (MACE-like) energy + forces on gfx1250.
// Live-path pruning: only l3==0 paths of interaction-2 feed the energy and all
// live CG tensors are diagonal. Dense 64-ch GEMMs run on
// v_wmma_f32_16x16x32_bf16 with an error-compensated bf16 hi/lo split
// (~fp32 accuracy); weights staged to LDS with global_load_async_to_lds_b128.
// Working set (~150 MB) fits in the 192 MB L2 -> edge-stream bound.
// ---------------------------------------------------------------------------

#define E_EDGES  160000
#define NN_NODES 10000
#define PI_F     3.14159265358979f
#define S3   1.73205081f
#define S5   2.23606798f
#define S7   2.64575131f
#define S15  3.87298335f
#define S42  6.48074069f
#define S70  8.36660027f
#define S105 10.24695077f
#define INV_NEIGH 0.0625f      // 1/AVG_NEIGH
#define BESSEL_C  0.63245553f  // sqrt(2/R_MAX)

typedef __attribute__((ext_vector_type(16))) __bf16 v16bf;
typedef __attribute__((ext_vector_type(8)))  float  v8f;

// ------------------------------ helpers ------------------------------------

__device__ __forceinline__ float bf2f(unsigned short h) {
  union { unsigned u; float f; } v; v.u = ((unsigned)h) << 16; return v.f;
}
__device__ __forceinline__ unsigned short f2bf(float x) {
  union { float f; unsigned u; } v; v.f = x;
  unsigned r = v.u + 0x7FFFu + ((v.u >> 16) & 1u);  // RNE
  return (unsigned short)(r >> 16);
}
__device__ __forceinline__ void split_bf(float x, __bf16& hi, __bf16& lo) {
  unsigned short h = f2bf(x);
  hi = __builtin_bit_cast(__bf16, h);
  unsigned short l = f2bf(x - bf2f(h));
  lo = __builtin_bit_cast(__bf16, l);
}
__device__ __forceinline__ float silu_f(float z) {
  float s = 1.f / (1.f + __expf(-z)); return z * s;
}
__device__ __forceinline__ float dsilu_f(float z) {
  float s = 1.f / (1.f + __expf(-z)); return s * (1.f + z * (1.f - s));
}

// --------------------- gfx1250 async copy to LDS ---------------------------
// Generic shared pointers carry the LDS offset in their low 32 bits
// (see ISA 10.2, LDS aperture).

__device__ __forceinline__ void async_copy_b32(const float* g, float* lds) {
  asm volatile("global_load_async_to_lds_b32 %0, %1, off"
               :: "v"((unsigned)(unsigned long long)lds),
                  "v"((unsigned long long)g)
               : "memory");
}
__device__ __forceinline__ void async_copy_b128(const float* g, float* lds) {
  asm volatile("global_load_async_to_lds_b128 %0, %1, off"
               :: "v"((unsigned)(unsigned long long)lds),
                  "v"((unsigned long long)g)
               : "memory");
}
__device__ __forceinline__ void wait_async() {
  asm volatile("s_wait_asynccnt 0x0" ::: "memory");
}

// ------------------------------ zero ---------------------------------------

__global__ void k_zero(float* __restrict__ p, long n) {
  long i = (long)blockIdx.x * blockDim.x + threadIdx.x;
  if (i < n) p[i] = 0.f;
}

// ------------------------------ edge geometry ------------------------------
// geo[e][4] = (ux,uy,uz,r); Yb[e][16] = real sph harmonics l=0..3 (ref order);
// feat[e][8] = bessel * polynomial cutoff.

__global__ void k_edge_geom(const float* __restrict__ pos,
                            const float* __restrict__ shifts,
                            const int* __restrict__ eidx,
                            float* __restrict__ geo,
                            float* __restrict__ Yb,
                            float* __restrict__ feat, int E) {
  int e = blockIdx.x * blockDim.x + threadIdx.x;
  if (e >= E) return;
  int s = eidx[e], rc = eidx[E + e];
  float vx = pos[rc*3+0] - pos[s*3+0] + shifts[e*3+0];
  float vy = pos[rc*3+1] - pos[s*3+1] + shifts[e*3+1];
  float vz = pos[rc*3+2] - pos[s*3+2] + shifts[e*3+2];
  float r = sqrtf(vx*vx + vy*vy + vz*vz + 1e-12f);
  float inv = 1.f / r;
  float x = vx*inv, y = vy*inv, z = vz*inv;
  geo[e*4+0]=x; geo[e*4+1]=y; geo[e*4+2]=z; geo[e*4+3]=r;
  float x2=x*x, y2=y*y, z2=z*z;
  float Y[16];
  Y[0]=1.f;
  Y[1]=S3*y;  Y[2]=S3*z;  Y[3]=S3*x;
  Y[4]=S15*x*y; Y[5]=S15*y*z; Y[6]=0.5f*S5*(3.f*z2-1.f);
  Y[7]=S15*x*z; Y[8]=0.5f*S15*(x2-y2);
  Y[9] =0.25f*S70*y*(3.f*x2-y2);
  Y[10]=S105*x*y*z;
  Y[11]=0.25f*S42*y*(5.f*z2-1.f);
  Y[12]=0.5f*S7*z*(5.f*z2-3.f);
  Y[13]=0.25f*S42*x*(5.f*z2-1.f);
  Y[14]=0.5f*S105*z*(x2-y2);
  Y[15]=0.25f*S70*x*(x2-3.f*y2);
#pragma unroll
  for (int i = 0; i < 16; ++i) Yb[(long)e*16 + i] = Y[i];
  float u = r * 0.2f;
  float fc = 0.f;
  if (u < 1.f) {
    float u2=u*u, u4=u2*u2, u5=u4*u, u6=u5*u, u7=u6*u;
    fc = 1.f - 21.f*u5 + 35.f*u6 - 15.f*u7;   // p=5 polynomial cutoff
  }
#pragma unroll
  for (int nb = 0; nb < 8; ++nb) {
    float w = (float)(nb+1) * PI_F * 0.2f;
    feat[(long)e*8 + nb] = BESSEL_C * sinf(w*r) * inv * fc;
  }
}

// ------------------------------ embedding ----------------------------------

__global__ void k_embed(const float* __restrict__ attrs,
                        const float* __restrict__ w_emb,
                        float* __restrict__ h0,
                        int* __restrict__ species, int Nn) {
  int t = blockIdx.x * blockDim.x + threadIdx.x;
  if (t >= Nn*64) return;
  int n = t >> 6, c = t & 63;
  int sp = 0; float acc = 0.f;
  for (int j = 0; j < 10; ++j) {
    float a = attrs[n*10 + j];
    acc += a * w_emb[j*64 + c];
    if (a > 0.5f) sp = j;
  }
  h0[t] = acc;
  if (c == 0) species[n] = sp;
}

// ------------------------------ WMMA GEMM ----------------------------------
// Specialized K=64, N=64 (all call sites). C[M,64] (=|+=) A[M,64] x B[64,64]
// (transB: B stored [64,64] accessed transposed).
// Block = 256 threads (8 waves); wave computes a 16x64 stripe (4 tiles).
// Weight tile staged to LDS with async b128 copies, pre-split to bf16 hi/lo.
// bf16 hi/lo split: acc += Ah*Bh + Ah*Bl + Al*Bh (~fp32).  24 static WMMAs.

__global__ void __launch_bounds__(256)
k_gemm(const float* __restrict__ A, const float* __restrict__ B,
       float* __restrict__ C, int M, int transB, int accumulate) {
  __shared__ alignas(16) float  Bf[4096];
  __shared__ __bf16 Bh[4096];
  __shared__ __bf16 Bl[4096];
  int lane = threadIdx.x & 31, wave = threadIdx.x >> 5;
  int half = lane >> 4, rr = lane & 15;

  // stage the 64x64 weight tile (16 KB) with async copies
#pragma unroll
  for (int j = 0; j < 4; ++j) {
    int t4 = (threadIdx.x + j*256) * 4;
    async_copy_b128(&B[t4], &Bf[t4]);
  }
  wait_async();
  __syncthreads();
#pragma unroll
  for (int j = 0; j < 16; ++j) {
    int idx = threadIdx.x + j*256;
    __bf16 h_, l_;
    split_bf(Bf[idx], h_, l_);
    Bh[idx] = h_; Bl[idx] = l_;
  }
  __syncthreads();

  int row0 = blockIdx.x * 128 + wave * 16;
  int arow = row0 + rr;
  v8f acc[4] = {{}, {}, {}, {}};
#pragma unroll
  for (int k0i = 0; k0i < 2; ++k0i) {
    int k0 = k0i * 32;
    v16bf ah, al;
#pragma unroll
    for (int i = 0; i < 16; ++i) {
      int kk = k0 + (i & 7) + half*8 + (i >> 3)*16;   // A 16x32 frag layout
      float a = (arow < M) ? A[(long)arow*64 + kk] : 0.f;
      __bf16 h_, l_;
      split_bf(a, h_, l_); ah[i] = h_; al[i] = l_;
    }
#pragma unroll
    for (int nt = 0; nt < 4; ++nt) {
      v16bf bh, bl;
#pragma unroll
      for (int i = 0; i < 16; ++i) {
        int kk = k0 + (i & 7) + half*8 + (i >> 3)*16; // B 32x16 frag layout
        int col = nt*16 + rr;
        int idx = transB ? col*64 + kk : kk*64 + col;
        bh[i] = Bh[idx]; bl[i] = Bl[idx];
      }
      acc[nt] = __builtin_amdgcn_wmma_f32_16x16x32_bf16(false, ah, false, bh, (short)0, acc[nt], false, false);
      acc[nt] = __builtin_amdgcn_wmma_f32_16x16x32_bf16(false, ah, false, bl, (short)0, acc[nt], false, false);
      acc[nt] = __builtin_amdgcn_wmma_f32_16x16x32_bf16(false, al, false, bh, (short)0, acc[nt], false, false);
    }
  }
#pragma unroll
  for (int nt = 0; nt < 4; ++nt) {
#pragma unroll
    for (int v = 0; v < 8; ++v) {
      int orow = row0 + half*8 + v;
      int ocol = nt*16 + rr;
      if (orow < M) {
        long idx = (long)orow*64 + ocol;
        C[idx] = (accumulate ? C[idx] : 0.f) + acc[nt][v];
      }
    }
  }
}

// -------------------- species-indexed self connection ----------------------
// out[n,d] += sum_c in[n,c] * W[sp(n)][c,d]   (trans: W[sp(n)][d,c])

__global__ void k_selfconn(const float* __restrict__ in,
                           const float* __restrict__ W,
                           float* __restrict__ out,
                           const int* __restrict__ species,
                           int Nn, int transW) {
  int t = blockIdx.x * blockDim.x + threadIdx.x;
  if (t >= Nn*64) return;
  int n = t >> 6, d = t & 63;
  const float* Wn = W + (long)species[n]*4096;
  float acc = 0.f;
  for (int c = 0; c < 64; ++c)
    acc += in[(long)n*64 + c] * (transW ? Wn[d*64 + c] : Wn[c*64 + d]);
  out[t] += acc;
}

// ------------------------ fused radial-MLP scratch -------------------------

struct RadScratch {
  float feat[8];
  float z1[64], a1[64];
  float z2[64], a2[64];
  float z3[64], a3[64];
  float g3[64], g2[64], g1[64];
  float gfeat[8];
  float gY[16];
};

__device__ void radial_fwd(int lane, RadScratch& W,
                           const float* __restrict__ W1,
                           const float* __restrict__ W2,
                           const float* __restrict__ W3) {
  for (int t = lane; t < 64; t += 32) {
    float z = 0.f;
#pragma unroll
    for (int i = 0; i < 8; ++i) z += W.feat[i] * W1[i*64 + t];
    W.z1[t] = z; W.a1[t] = silu_f(z);
  }
  __syncthreads();
  for (int t = lane; t < 64; t += 32) {
    float z = 0.f;
    for (int j = 0; j < 64; ++j) z += W.a1[j] * W2[j*64 + t];
    W.z2[t] = z; W.a2[t] = silu_f(z);
  }
  __syncthreads();
  for (int t = lane; t < 64; t += 32) {
    float z = 0.f;
    for (int j = 0; j < 64; ++j) z += W.a2[j] * W3[j*64 + t];
    W.z3[t] = z; W.a3[t] = silu_f(z);
  }
  __syncthreads();
}

// On entry W.g3 = dL/d a3 (accumulated). Produces W.gfeat.
__device__ void radial_bwd(int lane, RadScratch& W,
                           const float* __restrict__ W1,
                           const float* __restrict__ W2,
                           const float* __restrict__ W3) {
  for (int t = lane; t < 64; t += 32) W.g3[t] *= dsilu_f(W.z3[t]);
  __syncthreads();
  for (int t = lane; t < 64; t += 32) {
    float acc = 0.f;
    for (int k = 0; k < 64; ++k) acc += W3[t*64 + k] * W.g3[k];
    W.g2[t] = acc * dsilu_f(W.z2[t]);
  }
  __syncthreads();
  for (int t = lane; t < 64; t += 32) {
    float acc = 0.f;
    for (int k = 0; k < 64; ++k) acc += W2[t*64 + k] * W.g2[k];
    W.g1[t] = acc * dsilu_f(W.z1[t]);
  }
  __syncthreads();
  if (lane < 8) {
    float acc = 0.f;
    for (int j = 0; j < 64; ++j) acc += W1[lane*64 + j] * W.g1[j];
    W.gfeat[lane] = acc;
  }
  __syncthreads();
}

// -------------------- geometry backward (one lane) -------------------------
// grad_vec = u*dE/dr + (I - uu^T)/r * (dY/du)^T gY ; forces scatter.

__device__ void geom_force(const float* gY, const float* gfeat,
                           float x, float y, float z, float r,
                           float* __restrict__ F, int snd, int rcv) {
  float u = r * 0.2f;
  float gr = 0.f;
  if (u < 1.f) {
    float u2=u*u, u4=u2*u2, u5=u4*u, u6=u5*u, u7=u6*u;
    float fc  = 1.f - 21.f*u5 + 35.f*u6 - 15.f*u7;
    float dfc = (-105.f*u4 + 210.f*u5 - 105.f*u6) * 0.2f;
    float invr = 1.f / r;
    for (int nb = 0; nb < 8; ++nb) {
      float w = (float)(nb+1) * PI_F * 0.2f;
      float sn = sinf(w*r), cs = cosf(w*r);
      float b  = BESSEL_C * sn * invr;
      float db = BESSEL_C * (w*cs*invr - sn*invr*invr);
      gr += gfeat[nb] * (db*fc + b*dfc);
    }
  }
  float x2=x*x, y2=y*y, z2=z*z;
  float gx = S3*gY[3] + S15*y*gY[4] + S15*z*gY[7] + S15*x*gY[8]
           + 1.5f*S70*x*y*gY[9] + S105*y*z*gY[10]
           + 0.25f*S42*(5.f*z2-1.f)*gY[13] + S105*x*z*gY[14]
           + 0.75f*S70*(x2-y2)*gY[15];
  float gy = S3*gY[1] + S15*x*gY[4] + S15*z*gY[5] - S15*y*gY[8]
           + 0.75f*S70*(x2-y2)*gY[9] + S105*x*z*gY[10]
           + 0.25f*S42*(5.f*z2-1.f)*gY[11] - S105*y*z*gY[14]
           - 1.5f*S70*x*y*gY[15];
  float gz = S3*gY[2] + S15*y*gY[5] + 3.f*S5*z*gY[6] + S15*x*gY[7]
           + S105*x*y*gY[10] + 2.5f*S42*y*z*gY[11]
           + S7*(7.5f*z2-1.5f)*gY[12] + 2.5f*S42*x*z*gY[13]
           + 0.5f*S105*(x2-y2)*gY[14];
  float invr = 1.f / r;
  float dotg = x*gx + y*gy + z*gz;
  float vx = x*gr + (gx - x*dotg)*invr;
  float vy = y*gr + (gy - y*dotg)*invr;
  float vz = z*gr + (gz - z*dotg)*invr;
  // vec = pos[rcv]-pos[snd]; F = -dE/dpos
  atomicAdd(&F[snd*3+0],  vx); atomicAdd(&F[snd*3+1],  vy); atomicAdd(&F[snd*3+2],  vz);
  atomicAdd(&F[rcv*3+0], -vx); atomicAdd(&F[rcv*3+1], -vy); atomicAdd(&F[rcv*3+2], -vz);
}

// ---------------------- per-edge feature staging ---------------------------

__device__ __forceinline__ void stage_feat(int lane, RadScratch& W,
                                           const float* __restrict__ feat,
                                           long ee, bool valid) {
  if (lane < 8) {
    if (valid) async_copy_b32(&feat[ee*8 + lane], &W.feat[lane]);
    else       W.feat[lane] = 0.f;
  }
  wait_async();
  __syncthreads();
}

// ---------------------- interaction 1 message forward ----------------------
// paths (l,0,l), l=0..2; CG diagonal -> msg[c,k] = Y_l[k]*h0up[c]*R[p,c]/sqrt(2l+1)

__global__ void __launch_bounds__(128)
k_msg_fwd_l1(const float* __restrict__ Yb, const float* __restrict__ feat,
             const int* __restrict__ eidx, const float* __restrict__ hup0,
             const float* __restrict__ W1, const float* __restrict__ W2,
             const float* __restrict__ W3, const float* __restrict__ W4,
             float* __restrict__ agg, int E, int Nn) {
  __shared__ RadScratch S[4];
  int wv = threadIdx.x >> 5, lane = threadIdx.x & 31;
  RadScratch& W = S[wv];
  int e = blockIdx.x * 4 + wv;
  bool valid = e < E;
  int ee = valid ? e : 0;
  int snd = eidx[ee], rcv = eidx[E + ee];
  if (valid) __builtin_prefetch(hup0 + (long)snd*64, 0, 3);   // global_prefetch_b8
  stage_feat(lane, W, feat, ee, valid);
  radial_fwd(lane, W, W1, W2, W3);
  int c0 = lane, c1 = lane + 32;
  float hj0 = valid ? hup0[(long)snd*64 + c0] : 0.f;
  float hj1 = valid ? hup0[(long)snd*64 + c1] : 0.f;
  const float INV_S[3] = {1.f, 0.57735027f, 0.44721360f};
  const int   YO[3]    = {0, 1, 4};
  for (int p = 0; p < 3; ++p) {
    float R0 = 0.f, R1 = 0.f;
    for (int j = 0; j < 64; ++j) {
      float a = W.a3[j];
      R0 += a * W4[j*192 + p*64 + c0];
      R1 += a * W4[j*192 + p*64 + c1];
    }
    if (!valid) continue;
    float m0 = hj0*R0*INV_S[p]*INV_NEIGH, m1 = hj1*R1*INV_S[p]*INV_NEIGH;
    int kd = 2*p + 1;
    long pb = (p == 0) ? 0L : (p == 1 ? (long)Nn*64 : (long)Nn*256);
    for (int k = 0; k < kd; ++k) {
      float yv = Yb[(long)ee*16 + YO[p] + k];
      atomicAdd(&agg[pb + ((long)rcv*kd + k)*64 + c0], yv*m0);
      atomicAdd(&agg[pb + ((long)rcv*kd + k)*64 + c1], yv*m1);
    }
  }
}

// ---------------------- interaction 2 message forward ----------------------
// live paths (l,l,0): msg[c] = (sum_j Y_l[j]*hup1_l[c,j]) * R[p,c] / sqrt(2l+1)

__global__ void __launch_bounds__(128)
k_msg_fwd_l2(const float* __restrict__ Yb, const float* __restrict__ feat,
             const int* __restrict__ eidx, const float* __restrict__ hup1,
             const float* __restrict__ W1, const float* __restrict__ W2,
             const float* __restrict__ W3, const float* __restrict__ W4,
             float* __restrict__ agg1, int E, int Nn) {
  __shared__ RadScratch S[4];
  int wv = threadIdx.x >> 5, lane = threadIdx.x & 31;
  RadScratch& W = S[wv];
  int e = blockIdx.x * 4 + wv;
  bool valid = e < E;
  int ee = valid ? e : 0;
  int snd = eidx[ee], rcv = eidx[E + ee];
  if (valid) __builtin_prefetch(hup1 + (long)snd*64, 0, 3);
  stage_feat(lane, W, feat, ee, valid);
  radial_fwd(lane, W, W1, W2, W3);
  int c0 = lane, c1 = lane + 32;
  const float INV_S[3] = {1.f, 0.57735027f, 0.44721360f};
  const int   YO[3]    = {0, 1, 4};
  const int   PCOL[3]  = {0, 256, 576};   // live PATHS1 slots 0,4,9 in r1_w4
  for (int q = 0; q < 3; ++q) {
    float R0 = 0.f, R1 = 0.f;
    for (int j = 0; j < 64; ++j) {
      float a = W.a3[j];
      R0 += a * W4[j*832 + PCOL[q] + c0];
      R1 += a * W4[j*832 + PCOL[q] + c1];
    }
    if (!valid) continue;
    int kd = 2*q + 1;
    long hb = (q == 0) ? 0L : (q == 1 ? (long)Nn*64 : (long)Nn*256);
    float d0 = 0.f, d1 = 0.f;
    for (int j = 0; j < kd; ++j) {
      float yv = Yb[(long)ee*16 + YO[q] + j];
      d0 += yv * hup1[hb + ((long)snd*kd + j)*64 + c0];
      d1 += yv * hup1[hb + ((long)snd*kd + j)*64 + c1];
    }
    float m0 = d0*R0*INV_S[q]*INV_NEIGH, m1 = d1*R1*INV_S[q]*INV_NEIGH;
    atomicAdd(&agg1[(long)q*Nn*64 + (long)rcv*64 + c0], m0);
    atomicAdd(&agg1[(long)q*Nn*64 + (long)rcv*64 + c1], m1);
  }
}

// ------------------------------ readouts -----------------------------------

__global__ void k_readout(const int* __restrict__ species,
                          const int* __restrict__ batch,
                          const float* __restrict__ atomic_e,
                          const float* __restrict__ h1_0,
                          const float* __restrict__ h2_0,
                          const float* __restrict__ w_r0,
                          const float* __restrict__ w_m1,
                          const float* __restrict__ w_m2,
                          float* __restrict__ eacc, int Nn) {
  int n = blockIdx.x * blockDim.x + threadIdx.x;
  if (n >= Nn) return;
  float e0 = atomic_e[species[n]];
  float e1 = 0.f;
  for (int c = 0; c < 64; ++c) e1 += h1_0[(long)n*64 + c] * w_r0[c];
  float e2 = 0.f;
  for (int i = 0; i < 16; ++i) {
    float z = 0.f;
    for (int c = 0; c < 64; ++c) z += h2_0[(long)n*64 + c] * w_m1[c*16 + i];
    e2 += silu_f(z) * w_m2[i];
  }
  int g = batch[n];
  atomicAdd(&eacc[g*3+0], e0);
  atomicAdd(&eacc[g*3+1], e1);
  atomicAdd(&eacc[g*3+2], e2);
}

__global__ void k_finalize(const float* __restrict__ eacc, float* __restrict__ out) {
  int g = threadIdx.x;
  if (g >= 16) return;
  float a = eacc[g*3+0], b = eacc[g*3+1], c = eacc[g*3+2];
  out[g] = a + b + c;
  out[16 + g*3 + 0] = a;
  out[16 + g*3 + 1] = b;
  out[16 + g*3 + 2] = c;
}

// ------------------------------ e2 backward --------------------------------

__global__ void k_e2_bwd(const float* __restrict__ h2_0,
                         const float* __restrict__ w_m1,
                         const float* __restrict__ w_m2,
                         float* __restrict__ g_h2, int Nn) {
  int n = blockIdx.x * blockDim.x + threadIdx.x;
  if (n >= Nn) return;
  float gt[16];
  for (int i = 0; i < 16; ++i) {
    float z = 0.f;
    for (int c = 0; c < 64; ++c) z += h2_0[(long)n*64 + c] * w_m1[c*16 + i];
    gt[i] = w_m2[i] * dsilu_f(z);
  }
  for (int c = 0; c < 64; ++c) {
    float acc = 0.f;
    for (int i = 0; i < 16; ++i) acc += w_m1[c*16 + i] * gt[i];
    g_h2[(long)n*64 + c] = acc;
  }
}

__global__ void k_addvec(float* __restrict__ g, const float* __restrict__ v, int Nn) {
  int t = blockIdx.x * blockDim.x + threadIdx.x;
  if (t < Nn*64) g[t] += v[t & 63];
}

// ---------------------- interaction 2 message backward ---------------------

__global__ void __launch_bounds__(128)
k_msg_bwd_l2(const float* __restrict__ geo, const float* __restrict__ Yb,
             const float* __restrict__ feat, const int* __restrict__ eidx,
             const float* __restrict__ hup1, const float* __restrict__ gagg1,
             const float* __restrict__ W1, const float* __restrict__ W2,
             const float* __restrict__ W3, const float* __restrict__ W4,
             float* __restrict__ ghup1, float* __restrict__ F, int E, int Nn) {
  __shared__ RadScratch S[4];
  int wv = threadIdx.x >> 5, lane = threadIdx.x & 31;
  RadScratch& W = S[wv];
  int e = blockIdx.x * 4 + wv;
  bool valid = e < E;
  int ee = valid ? e : 0;
  int snd = eidx[ee], rcv = eidx[E + ee];
  if (valid) __builtin_prefetch(gagg1 + (long)rcv*64, 0, 3);
  if (lane < 16) W.gY[lane] = 0.f;
  for (int t = lane; t < 64; t += 32) W.g3[t] = 0.f;
  stage_feat(lane, W, feat, ee, valid);
  radial_fwd(lane, W, W1, W2, W3);
  int c0 = lane, c1 = lane + 32;
  const float INV_S[3] = {1.f, 0.57735027f, 0.44721360f};
  const int   YO[3]    = {0, 1, 4};
  const int   PCOL[3]  = {0, 256, 576};
  float gR[3][2];
  for (int q = 0; q < 3; ++q) {
    float R0 = 0.f, R1 = 0.f;
    for (int j = 0; j < 64; ++j) {
      float a = W.a3[j];
      R0 += a * W4[j*832 + PCOL[q] + c0];
      R1 += a * W4[j*832 + PCOL[q] + c1];
    }
    float ga0 = 0.f, ga1 = 0.f;
    if (valid) {
      ga0 = gagg1[(long)q*Nn*64 + (long)rcv*64 + c0] * INV_S[q] * INV_NEIGH;
      ga1 = gagg1[(long)q*Nn*64 + (long)rcv*64 + c1] * INV_S[q] * INV_NEIGH;
    }
    int kd = 2*q + 1;
    long hb = (q == 0) ? 0L : (q == 1 ? (long)Nn*64 : (long)Nn*256);
    float d0 = 0.f, d1 = 0.f;
    for (int j = 0; j < kd; ++j) {
      float yv = Yb[(long)ee*16 + YO[q] + j];
      float hj0 = valid ? hup1[hb + ((long)snd*kd + j)*64 + c0] : 0.f;
      float hj1 = valid ? hup1[hb + ((long)snd*kd + j)*64 + c1] : 0.f;
      d0 += yv*hj0; d1 += yv*hj1;
      if (valid) {
        atomicAdd(&ghup1[hb + ((long)snd*kd + j)*64 + c0], ga0*yv*R0);
        atomicAdd(&ghup1[hb + ((long)snd*kd + j)*64 + c1], ga1*yv*R1);
        atomicAdd(&W.gY[YO[q] + j], ga0*hj0*R0 + ga1*hj1*R1);
      }
    }
    gR[q][0] = ga0 * d0;
    gR[q][1] = ga1 * d1;
  }
  __syncthreads();
  // dL/da3 accumulation across channels
  for (int j = 0; j < 64; ++j) {
    float part = 0.f;
    for (int q = 0; q < 3; ++q)
      part += W4[j*832 + PCOL[q] + c0]*gR[q][0] + W4[j*832 + PCOL[q] + c1]*gR[q][1];
    atomicAdd(&W.g3[j], part);
  }
  __syncthreads();
  radial_bwd(lane, W, W1, W2, W3);
  if (lane == 0 && valid) {
    float x = geo[(long)ee*4+0], y = geo[(long)ee*4+1];
    float z = geo[(long)ee*4+2], r = geo[(long)ee*4+3];
    geom_force(W.gY, W.gfeat, x, y, z, r, F, snd, rcv);
  }
}

// ---------------------- interaction 1 message backward ---------------------
// hup0 is position-independent -> only grad_Y and grad_R matter here.

__global__ void __launch_bounds__(128)
k_msg_bwd_l1(const float* __restrict__ geo, const float* __restrict__ Yb,
             const float* __restrict__ feat, const int* __restrict__ eidx,
             const float* __restrict__ hup0, const float* __restrict__ gagg0,
             const float* __restrict__ W1, const float* __restrict__ W2,
             const float* __restrict__ W3, const float* __restrict__ W4,
             float* __restrict__ F, int E, int Nn) {
  __shared__ RadScratch S[4];
  int wv = threadIdx.x >> 5, lane = threadIdx.x & 31;
  RadScratch& W = S[wv];
  int e = blockIdx.x * 4 + wv;
  bool valid = e < E;
  int ee = valid ? e : 0;
  int snd = eidx[ee], rcv = eidx[E + ee];
  if (valid) __builtin_prefetch(gagg0 + (long)rcv*64, 0, 3);
  if (lane < 16) W.gY[lane] = 0.f;
  for (int t = lane; t < 64; t += 32) W.g3[t] = 0.f;
  stage_feat(lane, W, feat, ee, valid);
  radial_fwd(lane, W, W1, W2, W3);
  int c0 = lane, c1 = lane + 32;
  float hj0 = valid ? hup0[(long)snd*64 + c0] : 0.f;
  float hj1 = valid ? hup0[(long)snd*64 + c1] : 0.f;
  const float INV_S[3] = {1.f, 0.57735027f, 0.44721360f};
  const int   YO[3]    = {0, 1, 4};
  float gR[3][2];
  for (int p = 0; p < 3; ++p) {
    float R0 = 0.f, R1 = 0.f;
    for (int j = 0; j < 64; ++j) {
      float a = W.a3[j];
      R0 += a * W4[j*192 + p*64 + c0];
      R1 += a * W4[j*192 + p*64 + c1];
    }
    int kd = 2*p + 1;
    long pb = (p == 0) ? 0L : (p == 1 ? (long)Nn*64 : (long)Nn*256);
    float s0 = 0.f, s1 = 0.f;
    for (int k = 0; k < kd; ++k) {
      float yv = Yb[(long)ee*16 + YO[p] + k];
      float g0 = 0.f, g1 = 0.f;
      if (valid) {
        g0 = gagg0[pb + ((long)rcv*kd + k)*64 + c0] * INV_S[p] * INV_NEIGH;
        g1 = gagg0[pb + ((long)rcv*kd + k)*64 + c1] * INV_S[p] * INV_NEIGH;
      }
      s0 += yv*g0; s1 += yv*g1;
      atomicAdd(&W.gY[YO[p] + k], g0*hj0*R0 + g1*hj1*R1);
    }
    gR[p][0] = hj0 * s0;
    gR[p][1] = hj1 * s1;
  }
  __syncthreads();
  for (int j = 0; j < 64; ++j) {
    float part = 0.f;
    for (int p = 0; p < 3; ++p)
      part += W4[j*192 + p*64 + c0]*gR[p][0] + W4[j*192 + p*64 + c1]*gR[p][1];
    atomicAdd(&W.g3[j], part);
  }
  __syncthreads();
  radial_bwd(lane, W, W1, W2, W3);
  if (lane == 0 && valid) {
    float x = geo[(long)ee*4+0], y = geo[(long)ee*4+1];
    float z = geo[(long)ee*4+2], r = geo[(long)ee*4+3];
    geom_force(W.gY, W.gfeat, x, y, z, r, F, snd, rcv);
  }
}

// ------------------------------ launcher -----------------------------------

extern "C" void kernel_launch(void* const* d_in, const int* in_sizes, int n_in,
                              void* d_out, int out_size, void* d_ws, size_t ws_size,
                              hipStream_t stream) {
  const int E = E_EDGES, Nn = NN_NODES;
  // inputs (setup_inputs order)
  const float* positions = (const float*)d_in[0];
  const float* node_attrs= (const float*)d_in[1];
  const float* shifts    = (const float*)d_in[2];
  const float* atomic_e  = (const float*)d_in[3];
  const float* w_emb     = (const float*)d_in[4];
  const float* w_up0     = (const float*)d_in[5];
  const float* r0_w1     = (const float*)d_in[6];
  const float* r0_w2     = (const float*)d_in[7];
  const float* r0_w3     = (const float*)d_in[8];
  const float* r0_w4     = (const float*)d_in[9];
  const float* w_out0    = (const float*)d_in[10];
  const float* w_sc0     = (const float*)d_in[11];
  const float* w_r0      = (const float*)d_in[12];
  const float* w_up1     = (const float*)d_in[13];
  const float* r1_w1     = (const float*)d_in[14];
  const float* r1_w2     = (const float*)d_in[15];
  const float* r1_w3     = (const float*)d_in[16];
  const float* r1_w4     = (const float*)d_in[17];
  const float* w_out1    = (const float*)d_in[18];
  const float* w_sc1     = (const float*)d_in[19];
  const float* w_m1      = (const float*)d_in[20];
  const float* w_m2      = (const float*)d_in[21];
  const int*   edge_index= (const int*)  d_in[22];
  const int*   batch     = (const int*)  d_in[23];
  (void)in_sizes; (void)n_in; (void)out_size; (void)ws_size;

  float* out   = (float*)d_out;
  float* force = out + 64;   // total(16) + contributions(48), forces follow

  // workspace carve-out
  float* wsf = (float*)d_ws;
  long off = 0;
  auto alloc = [&](long n) { float* p = wsf + off; off += n; return p; };
  float* geo   = alloc(4L*E);
  float* Yb    = alloc(16L*E);
  float* featb = alloc(8L*E);
  float* h0    = alloc(64L*Nn);
  float* hup0  = alloc(64L*Nn);
  float* agg0  = alloc(576L*Nn);   // paths l=0,1,2 : (1+3+5)*64
  float* h1    = alloc(576L*Nn);   // l blocks at 0, 64N, 256N
  float* hup1  = alloc(576L*Nn);
  float* agg1  = alloc(192L*Nn);   // 3 live paths * 64
  float* h2    = alloc(64L*Nn);
  float* gh2   = alloc(64L*Nn);
  float* gagg1 = alloc(192L*Nn);
  float* ghup1 = alloc(576L*Nn);
  float* gh1   = alloc(576L*Nn);
  float* gagg0 = alloc(576L*Nn);
  float* eacc  = alloc(64);
  int* species = (int*)(wsf + off); off += Nn;
  const long WS_FLOATS = off;

  auto zero = [&](float* p, long n) {
    k_zero<<<dim3((unsigned)((n + 255) / 256)), 256, 0, stream>>>(p, n);
  };
  auto gemm = [&](const float* A, const float* B, float* C,
                  int M, int transB, int acc) {
    k_gemm<<<dim3((M + 127) / 128), 256, 0, stream>>>(A, B, C, M, transB, acc);
  };

  const long OFFL[3] = {0, 64L*Nn, 256L*Nn};
  const int  P2[3]   = {0, 4, 9};          // live PATHS1 indices (l3==0)

  zero(wsf, WS_FLOATS);
  zero(out, 64 + 3L*Nn);

  // -------- forward --------
  k_edge_geom<<<(E + 255)/256, 256, 0, stream>>>(positions, shifts, edge_index,
                                                 geo, Yb, featb, E);
  k_embed<<<(Nn*64 + 255)/256, 256, 0, stream>>>(node_attrs, w_emb, h0, species, Nn);
  gemm(h0, w_up0, hup0, Nn, 0, 0);
  k_msg_fwd_l1<<<(E + 3)/4, 128, 0, stream>>>(Yb, featb, edge_index, hup0,
                                              r0_w1, r0_w2, r0_w3, r0_w4, agg0, E, Nn);
  for (int p = 0; p < 3; ++p)
    gemm(agg0 + OFFL[p], w_out0 + p*4096, h1 + OFFL[p], Nn*(2*p+1), 0, 1);
  k_selfconn<<<(Nn*64 + 255)/256, 256, 0, stream>>>(h0, w_sc0, h1, species, Nn, 0);
  for (int l = 0; l < 3; ++l)
    gemm(h1 + OFFL[l], w_up1 + l*4096, hup1 + OFFL[l], Nn*(2*l+1), 0, 0);
  k_msg_fwd_l2<<<(E + 3)/4, 128, 0, stream>>>(Yb, featb, edge_index, hup1,
                                              r1_w1, r1_w2, r1_w3, r1_w4, agg1, E, Nn);
  for (int q = 0; q < 3; ++q)
    gemm(agg1 + (long)q*64*Nn, w_out1 + P2[q]*4096, h2, Nn, 0, 1);
  k_selfconn<<<(Nn*64 + 255)/256, 256, 0, stream>>>(h1, w_sc1, h2, species, Nn, 0);
  k_readout<<<(Nn + 255)/256, 256, 0, stream>>>(species, batch, atomic_e,
                                                h1, h2, w_r0, w_m1, w_m2, eacc, Nn);
  k_finalize<<<1, 32, 0, stream>>>(eacc, out);

  // -------- backward (forces) --------
  k_e2_bwd<<<(Nn + 255)/256, 256, 0, stream>>>(h2, w_m1, w_m2, gh2, Nn);
  for (int q = 0; q < 3; ++q)
    gemm(gh2, w_out1 + P2[q]*4096, gagg1 + (long)q*64*Nn, Nn, 1, 0);
  k_msg_bwd_l2<<<(E + 3)/4, 128, 0, stream>>>(geo, Yb, featb, edge_index, hup1, gagg1,
                                              r1_w1, r1_w2, r1_w3, r1_w4,
                                              ghup1, force, E, Nn);
  for (int l = 0; l < 3; ++l)
    gemm(ghup1 + OFFL[l], w_up1 + l*4096, gh1 + OFFL[l], Nn*(2*l+1), 1, 0);
  k_selfconn<<<(Nn*64 + 255)/256, 256, 0, stream>>>(gh2, w_sc1, gh1, species, Nn, 1);
  k_addvec<<<(Nn*64 + 255)/256, 256, 0, stream>>>(gh1, w_r0, Nn);
  for (int p = 0; p < 3; ++p)
    gemm(gh1 + OFFL[p], w_out0 + p*4096, gagg0 + OFFL[p], Nn*(2*p+1), 1, 0);
  k_msg_bwd_l1<<<(E + 3)/4, 128, 0, stream>>>(geo, Yb, featb, edge_index, hup0, gagg0,
                                              r0_w1, r0_w2, r0_w3, r0_w4,
                                              force, E, Nn);
}
